// ArtemisNet_67525475828407
// MI455X (gfx1250) — matrix operand order repaired and checked by reading, not compiled, not run on hardware
//
#include <hip/hip_runtime.h>
#include <hip/hip_bf16.h>
#include <stdint.h>

// GAT(128 -> 64x4 concat) -> GAT(256 -> 64) -> MLP(64->32->2) on gfx1250.
// Dense GEMMs use V_WMMA_F32_16X16X32_BF16 (bf16 in, f32 accum), one 16x64
// output tile per wave (A fragment reused across 4 WMMAs).
// Segment softmax via 3 edge passes with f32 global atomics.

#define HEADS 4
#define HIDC 64
#define INC 128
#define NEG_SLOPE 0.2f

typedef __attribute__((ext_vector_type(16))) __bf16 v16bf;
typedef __attribute__((ext_vector_type(8)))  float  v8f;

static inline int cdiv(int a, int b) { return (a + b - 1) / b; }

__device__ __forceinline__ unsigned short f32_bf16(float f) {
    unsigned int u = __float_as_uint(f);
    if ((u & 0x7FFFFFFFu) > 0x7F800000u) return (unsigned short)((u >> 16) | 0x0040u); // NaN
    unsigned int r = u + 0x7FFFu + ((u >> 16) & 1u);  // round-to-nearest-even
    return (unsigned short)(r >> 16);
}

__device__ __forceinline__ void atomicMaxF32(float* addr, float v) {
    // sign-aware monotone bit trick; addr pre-initialized to -inf
    if (v >= 0.f) atomicMax((int*)addr, __float_as_int(v));
    else          atomicMin((unsigned int*)addr, (unsigned int)__float_as_int(v));
}

__device__ __forceinline__ float lrelu(float v) { return v > 0.f ? v : v * NEG_SLOPE; }
__device__ __forceinline__ float elu1(float v)  { return v > 0.f ? v : (__expf(v) - 1.f); }

// ---------------------------------------------------------------- conversions
__global__ void k_f32_to_bf16(const float* __restrict__ src,
                              unsigned short* __restrict__ dst, int n) {
    int t = blockIdx.x * blockDim.x + threadIdx.x;
    if (t < n) dst[t] = f32_bf16(src[t]);
}

// B [K][Nc] f32 (row major)  ->  BT [Nc][K] bf16
__global__ void k_transpose_bf16(const float* __restrict__ B,
                                 unsigned short* __restrict__ BT,
                                 int K, int Nc) {
    int t = blockIdx.x * blockDim.x + threadIdx.x;
    if (t >= K * Nc) return;
    int k = t / Nc, n = t - k * Nc;
    BT[(size_t)n * K + k] = f32_bf16(B[t]);
}

__global__ void k_fill_f32(float* __restrict__ p, float v, int n) {
    int t = blockIdx.x * blockDim.x + threadIdx.x;
    if (t < n) p[t] = v;
}

// ---------------------------------------------------------------- WMMA GEMM
// C[M,Nc] = A[M,K] @ B[K,Nc]; A bf16 row-major, BT = B^T bf16 row-major.
// One 16x64 tile per wave: a single A fragment feeds 4 WMMAs per K-step.
// Requires Nc % 64 == 0. Fragment K order: contiguous k = 16*(lane/16) + s, same for A and B.
__global__ void k_gemm_bf16_wmma(const unsigned short* __restrict__ A,
                                 const unsigned short* __restrict__ BT,
                                 float* __restrict__ C,
                                 int M, int K, int Nc) {
    int wave = (int)((blockIdx.x * blockDim.x + threadIdx.x) >> 5);
    int lane = threadIdx.x & 31;
    int nBlk = Nc >> 6;                    // 64-wide column blocks
    int mTiles = (M + 15) >> 4;
    if (wave >= mTiles * nBlk) return;
    int mt = wave / nBlk, nb = wave - mt * nBlk;
    int m0 = mt << 4, n0 = nb << 6;

    int rowA = m0 + (lane & 15);
    if (rowA >= M) rowA = M - 1;           // clamp reads; stores are guarded
    int kbA = (lane >> 4) << 3;            // 0 | 8
    int kbB = (lane >> 4) << 4;            // 0 | 16

    const unsigned short* aRow = A + (size_t)rowA * K;
    const unsigned short* b0r = BT + (size_t)(n0 +  0 + (lane & 15)) * K;
    const unsigned short* b1r = BT + (size_t)(n0 + 16 + (lane & 15)) * K;
    const unsigned short* b2r = BT + (size_t)(n0 + 32 + (lane & 15)) * K;
    const unsigned short* b3r = BT + (size_t)(n0 + 48 + (lane & 15)) * K;

    v8f c0 = {}, c1 = {}, c2 = {}, c3 = {};
    for (int k0 = 0; k0 < K; k0 += 32) {
        union { v16bf v; uint4 u[2]; } a, b0, b1, b2, b3;
        a.u[0]  = *(const uint4*)(aRow + k0 + kbB);       // elems 0..7 : K=kbB..+7
        a.u[1]  = *(const uint4*)(aRow + k0 + kbB + 8);   // elems 8..15: K=kbB+8..+15
        b0.u[0] = *(const uint4*)(b0r + k0 + kbB);
        b0.u[1] = *(const uint4*)(b0r + k0 + kbB + 8);
        b1.u[0] = *(const uint4*)(b1r + k0 + kbB);
        b1.u[1] = *(const uint4*)(b1r + k0 + kbB + 8);
        b2.u[0] = *(const uint4*)(b2r + k0 + kbB);
        b2.u[1] = *(const uint4*)(b2r + k0 + kbB + 8);
        b3.u[0] = *(const uint4*)(b3r + k0 + kbB);
        b3.u[1] = *(const uint4*)(b3r + k0 + kbB + 8);
        c0 = __builtin_amdgcn_wmma_f32_16x16x32_bf16(false, a.v, false, b0.v,
                                                     (short)0, c0, false, false);
        c1 = __builtin_amdgcn_wmma_f32_16x16x32_bf16(false, a.v, false, b1.v,
                                                     (short)0, c1, false, false);
        c2 = __builtin_amdgcn_wmma_f32_16x16x32_bf16(false, a.v, false, b2.v,
                                                     (short)0, c2, false, false);
        c3 = __builtin_amdgcn_wmma_f32_16x16x32_bf16(false, a.v, false, b3.v,
                                                     (short)0, c3, false, false);
    }

    int col   = lane & 15;
    int rbase = m0 + ((lane >> 4) << 3);
#pragma unroll
    for (int r = 0; r < 8; ++r) {
        int row = rbase + r;
        if (row < M) {
            float* crow = C + (size_t)row * Nc + n0 + col;
            crow[ 0] = c0[r];
            crow[16] = c1[r];
            crow[32] = c2[r];
            crow[48] = c3[r];
        }
    }
}

// ---------------------------------------------------------------- attention
// aS[n,h] = sum_c Hf[n, h*C + c] * attS[h,c]   (same for aD)
__global__ void k_att_dots(const float* __restrict__ Hf,
                           const float* __restrict__ attS,
                           const float* __restrict__ attD,
                           float* __restrict__ aS, float* __restrict__ aD,
                           int N, int H, int C) {
    int t = blockIdx.x * blockDim.x + threadIdx.x;
    if (t >= N * H) return;
    int n = t / H, h = t - n * H;
    const float* row = Hf + (size_t)n * H * C + (size_t)h * C;
    float s = 0.f, d = 0.f;
    for (int c = 0; c < C; ++c) {
        float v = row[c];
        s = fmaf(v, attS[h * C + c], s);
        d = fmaf(v, attD[h * C + c], d);
    }
    aS[t] = s; aD[t] = d;
}

// ---------------------------------------------------------------- edge passes
__global__ void k_edge_max(const int* __restrict__ src, const int* __restrict__ dst,
                           int E, int N, int H,
                           const float* __restrict__ aS, const float* __restrict__ aD,
                           float* __restrict__ m) {
    int e = blockIdx.x * blockDim.x + threadIdx.x;
    if (e >= E + N) return;
    int s, d;
    if (e < E) { s = src[e]; d = dst[e]; } else { s = d = e - E; }   // self loops
    for (int h = 0; h < H; ++h) {
        float v = lrelu(aS[s * H + h] + aD[d * H + h]);
        atomicMaxF32(&m[d * H + h], v);
    }
}

__global__ void k_edge_expsum(const int* __restrict__ src, const int* __restrict__ dst,
                              int E, int N, int H,
                              const float* __restrict__ aS, const float* __restrict__ aD,
                              const float* __restrict__ m,
                              float* __restrict__ ex, float* __restrict__ denom) {
    int e = blockIdx.x * blockDim.x + threadIdx.x;
    if (e >= E + N) return;
    int s, d;
    if (e < E) { s = src[e]; d = dst[e]; } else { s = d = e - E; }
    for (int h = 0; h < H; ++h) {
        float v  = lrelu(aS[s * H + h] + aD[d * H + h]);
        float mm = m[d * H + h];
        if (!(mm > -1e38f)) mm = 0.f;                 // isfinite fixup
        float x = __expf(v - mm);
        ex[(size_t)e * H + h] = x;
        atomicAdd(&denom[d * H + h], x);
    }
}

// Layer 1 messages: H=4 heads x 64 ch = 256 floats; one wave per edge.
__global__ void k_edge_scatter256(const int* __restrict__ src, const int* __restrict__ dst,
                                  int E, int N,
                                  const float* __restrict__ ex, const float* __restrict__ denom,
                                  const float* __restrict__ Hf, float* __restrict__ out) {
    int wave = (int)((blockIdx.x * blockDim.x + threadIdx.x) >> 5);
    int lane = threadIdx.x & 31;
    if (wave >= E + N) return;
    int s, d;
    if (wave < E) { s = src[wave]; d = dst[wave]; } else { s = d = wave - E; }
#pragma unroll
    for (int half = 0; half < 2; ++half) {
        int ch = 4 * lane + half * 128;
        int h  = ch >> 6;
        float alpha = ex[(size_t)wave * HEADS + h] / (denom[d * HEADS + h] + 1e-16f);
        float4 v = *(const float4*)(Hf + (size_t)s * 256 + ch);
        float* o = out + (size_t)d * 256 + ch;
        atomicAdd(o + 0, v.x * alpha);
        atomicAdd(o + 1, v.y * alpha);
        atomicAdd(o + 2, v.z * alpha);
        atomicAdd(o + 3, v.w * alpha);
    }
}

// Layer 2 messages: 1 head x 64 ch; one wave per edge, float2 per lane.
__global__ void k_edge_scatter64(const int* __restrict__ src, const int* __restrict__ dst,
                                 int E, int N,
                                 const float* __restrict__ ex, const float* __restrict__ denom,
                                 const float* __restrict__ Hf, float* __restrict__ out) {
    int wave = (int)((blockIdx.x * blockDim.x + threadIdx.x) >> 5);
    int lane = threadIdx.x & 31;
    if (wave >= E + N) return;
    int s, d;
    if (wave < E) { s = src[wave]; d = dst[wave]; } else { s = d = wave - E; }
    float alpha = ex[wave] / (denom[d] + 1e-16f);
    int ch = 2 * lane;
    float2 v = *(const float2*)(Hf + (size_t)s * 64 + ch);
    float* o = out + (size_t)d * 64 + ch;
    atomicAdd(o + 0, v.x * alpha);
    atomicAdd(o + 1, v.y * alpha);
}

// ---------------------------------------------------------------- finalize
// out := elu(out + bias);  optional bf16 copy for the next WMMA GEMM
__global__ void k_bias_elu(float* __restrict__ out, const float* __restrict__ bias,
                           unsigned short* __restrict__ out16, int N, int C) {
    int t = blockIdx.x * blockDim.x + threadIdx.x;
    if (t >= N * C) return;
    float v = elu1(out[t] + bias[t % C]);
    out[t] = v;
    if (out16) out16[t] = f32_bf16(v);
}

// ---------------------------------------------------------------- MLP head
// y = relu(z @ Wm1 + bm1) @ Wm2 + bm2 ;  z:[N,64], Wm1:[64,32], Wm2:[32,2]
// One wave per node: lane j owns hidden unit j; z broadcast via shuffles.
__global__ void k_mlp_head(const float* __restrict__ Z,
                           const float* __restrict__ Wm1, const float* __restrict__ bm1,
                           const float* __restrict__ Wm2, const float* __restrict__ bm2,
                           float* __restrict__ out, int N) {
    int wave = (int)((blockIdx.x * blockDim.x + threadIdx.x) >> 5);
    int lane = threadIdx.x & 31;
    if (wave >= N) return;
    const float* z = Z + (size_t)wave * 64;
    float z0 = z[lane], z1 = z[lane + 32];
    float hj = bm1[lane];
#pragma unroll
    for (int k = 0; k < 32; ++k)
        hj = fmaf(__shfl(z0, k, 32), Wm1[k * 32 + lane], hj);
#pragma unroll
    for (int k = 0; k < 32; ++k)
        hj = fmaf(__shfl(z1, k, 32), Wm1[(k + 32) * 32 + lane], hj);
    hj = fmaxf(hj, 0.f);
    float o0 = hj * Wm2[lane * 2 + 0];
    float o1 = hj * Wm2[lane * 2 + 1];
#pragma unroll
    for (int off = 16; off; off >>= 1) {
        o0 += __shfl_down(o0, off, 32);
        o1 += __shfl_down(o1, off, 32);
    }
    if (lane == 0) {
        out[(size_t)wave * 2 + 0] = o0 + bm2[0];
        out[(size_t)wave * 2 + 1] = o1 + bm2[1];
    }
}

// ---------------------------------------------------------------- launcher
extern "C" void kernel_launch(void* const* d_in, const int* in_sizes, int n_in,
                              void* d_out, int out_size, void* d_ws, size_t ws_size,
                              hipStream_t stream) {
    const float* x        = (const float*)d_in[0];
    const int*   eidx     = (const int*)d_in[1];
    const float* W1       = (const float*)d_in[2];
    const float* att_src1 = (const float*)d_in[3];
    const float* att_dst1 = (const float*)d_in[4];
    const float* b1       = (const float*)d_in[5];
    const float* W2       = (const float*)d_in[6];
    const float* att_src2 = (const float*)d_in[7];
    const float* att_dst2 = (const float*)d_in[8];
    const float* b2       = (const float*)d_in[9];
    const float* Wm1      = (const float*)d_in[10];
    const float* bm1      = (const float*)d_in[11];
    const float* Wm2      = (const float*)d_in[12];
    const float* bm2      = (const float*)d_in[13];
    (void)n_in; (void)out_size; (void)ws_size;

    const int N = in_sizes[0] / INC;        // 50000
    const int E = in_sizes[1] / 2;          // 800000
    const int Etot = E + N;                 // with self loops
    const int* srcI = eidx;
    const int* dstI = eidx + E;

    // ---- workspace carve-out (256B aligned)
    char* wsp = (char*)d_ws;
    size_t off = 0;
    auto carve = [&](size_t bytes) -> void* {
        void* p = wsp + off;
        off += (bytes + 255) & ~(size_t)255;
        return p;
    };
    unsigned short* xb    = (unsigned short*)carve((size_t)N * INC * 2);
    unsigned short* w1t   = (unsigned short*)carve((size_t)(HEADS * HIDC) * INC * 2);
    unsigned short* w2t   = (unsigned short*)carve((size_t)HIDC * (HEADS * HIDC) * 2);
    float*          h1    = (float*)carve((size_t)N * 256 * 4);
    unsigned short* h1b16 = (unsigned short*)carve((size_t)N * 256 * 2);
    float*          aS1   = (float*)carve((size_t)N * HEADS * 4);
    float*          aD1   = (float*)carve((size_t)N * HEADS * 4);
    float*          m1    = (float*)carve((size_t)N * HEADS * 4);
    float*          den1  = (float*)carve((size_t)N * HEADS * 4);
    float*          ex1   = (float*)carve((size_t)Etot * HEADS * 4);
    float*          out1  = (float*)carve((size_t)N * 256 * 4);
    float*          g2    = (float*)carve((size_t)N * HIDC * 4);
    float*          aS2   = (float*)carve((size_t)N * 4);
    float*          aD2   = (float*)carve((size_t)N * 4);
    float*          m2    = (float*)carve((size_t)N * 4);
    float*          den2  = (float*)carve((size_t)N * 4);
    float*          ex2   = (float*)carve((size_t)Etot * 4);
    float*          out2  = (float*)carve((size_t)N * HIDC * 4);

    const int TB = 256;
    const float NEG_INF = -__builtin_huge_valf();

    // ---- precision conversion / weight transpose
    k_f32_to_bf16<<<cdiv(N * INC, TB), TB, 0, stream>>>(x, xb, N * INC);
    k_transpose_bf16<<<cdiv(INC * 256, TB), TB, 0, stream>>>(W1, w1t, INC, 256);
    k_transpose_bf16<<<cdiv(256 * HIDC, TB), TB, 0, stream>>>(W2, w2t, 256, HIDC);

    // ---- layer 1: h1 = x @ W1  (bf16 WMMA, f32 accum)
    {
        int waves = cdiv(N, 16) * (256 / 64);
        k_gemm_bf16_wmma<<<cdiv(waves * 32, TB), TB, 0, stream>>>(xb, w1t, h1, N, INC, 256);
    }
    k_att_dots<<<cdiv(N * HEADS, TB), TB, 0, stream>>>(h1, att_src1, att_dst1, aS1, aD1,
                                                       N, HEADS, HIDC);
    k_fill_f32<<<cdiv(N * HEADS, TB), TB, 0, stream>>>(m1, NEG_INF, N * HEADS);
    k_fill_f32<<<cdiv(N * HEADS, TB), TB, 0, stream>>>(den1, 0.f, N * HEADS);
    k_fill_f32<<<cdiv(N * 256, TB), TB, 0, stream>>>(out1, 0.f, N * 256);

    k_edge_max<<<cdiv(Etot, TB), TB, 0, stream>>>(srcI, dstI, E, N, HEADS, aS1, aD1, m1);
    k_edge_expsum<<<cdiv(Etot, TB), TB, 0, stream>>>(srcI, dstI, E, N, HEADS,
                                                     aS1, aD1, m1, ex1, den1);
    k_edge_scatter256<<<cdiv(Etot * 32, TB), TB, 0, stream>>>(srcI, dstI, E, N,
                                                              ex1, den1, h1, out1);
    k_bias_elu<<<cdiv(N * 256, TB), TB, 0, stream>>>(out1, b1, h1b16, N, 256);

    // ---- layer 2: g2 = elu(out1) @ W2
    {
        int waves = cdiv(N, 16) * (HIDC / 64);
        k_gemm_bf16_wmma<<<cdiv(waves * 32, TB), TB, 0, stream>>>(h1b16, w2t, g2, N, 256, HIDC);
    }
    k_att_dots<<<cdiv(N, TB), TB, 0, stream>>>(g2, att_src2, att_dst2, aS2, aD2, N, 1, HIDC);
    k_fill_f32<<<cdiv(N, TB), TB, 0, stream>>>(m2, NEG_INF, N);
    k_fill_f32<<<cdiv(N, TB), TB, 0, stream>>>(den2, 0.f, N);
    k_fill_f32<<<cdiv(N * HIDC, TB), TB, 0, stream>>>(out2, 0.f, N * HIDC);

    k_edge_max<<<cdiv(Etot, TB), TB, 0, stream>>>(srcI, dstI, E, N, 1, aS2, aD2, m2);
    k_edge_expsum<<<cdiv(Etot, TB), TB, 0, stream>>>(srcI, dstI, E, N, 1,
                                                     aS2, aD2, m2, ex2, den2);
    k_edge_scatter64<<<cdiv(Etot * 32, TB), TB, 0, stream>>>(srcI, dstI, E, N,
                                                             ex2, den2, g2, out2);
    k_bias_elu<<<cdiv(N * HIDC, TB), TB, 0, stream>>>(out2, b2, nullptr, N, HIDC);

    // ---- MLP head -> d_out [N,2]
    k_mlp_head<<<cdiv(N * 32, TB), TB, 0, stream>>>(out2, Wm1, bm1, Wm2, bm2,
                                                    (float*)d_out, N);
}
